// YOLOV5TorchObjectDetector_41051297415526
// MI455X (gfx1250) — matrix operand order repaired and checked by reading, not compiled, not run on hardware
//
#include <hip/hip_runtime.h>
#include <stdint.h>

#define BATCH   16
#define NANCH   25200
#define NC      80
#define ROWLEN  85
#define PADN    32768      // next pow2 >= NANCH
#define KTOP    1024
#define MAXDET  300
#define CONF_T  0.25f
#define IOU_T   0.45f
#define MAX_WH  4096.0f
#define IMGW    640.0f
#define IMGH    640.0f
#define TILE    32         // anchors per wave in stage 1
#define SORT_TILE 2048     // elements per LDS sort block

typedef __attribute__((ext_vector_type(16))) _Float16 v16h;
typedef __attribute__((ext_vector_type(8)))  float    v8f;

// order-preserving float->uint mapping (ascending)
__device__ __forceinline__ uint32_t map_score(float f) {
  uint32_t u = __float_as_uint(f);
  return (u & 0x80000000u) ? ~u : (u | 0x80000000u);
}
__device__ __forceinline__ float unmap_score(uint32_t m) {
  uint32_t u = (m & 0x80000000u) ? (m ^ 0x80000000u) : ~m;
  return __uint_as_float(u);
}

// ---------------- Stage 1: score + sortable key ----------------------------
// Wave-cooperative: each wave stages its 32 rows (contiguous 32*85 floats)
// into LDS with lane-consecutive coalesced loads, then reduces from LDS.
// Bank-conflict free: row stride 85 = 21 (mod 64), gcd(21,64)=1.
__global__ __launch_bounds__(128)
void score_key_kernel(const float* __restrict__ pred,
                      uint64_t* __restrict__ keys) {
  __shared__ float rows[4][TILE * ROWLEN];          // 43520 B
  int waveInBlk = threadIdx.x >> 5;
  int lane = threadIdx.x & 31;
  int gwave = blockIdx.x * 4 + waveInBlk;           // all 4 waves same batch
  int b = gwave >> 10;                              // PADN/TILE = 1024 waves/batch
  int tile = gwave & 1023;
  int n0 = tile * TILE;

  int cnt = 0;
  const float* src = pred + ((size_t)b * NANCH + n0) * ROWLEN;
  if (n0 < NANCH) cnt = (NANCH - n0 < TILE) ? (NANCH - n0) : TILE;
  float* ld = rows[waveInBlk];
  int tot = cnt * ROWLEN;
  __builtin_prefetch(src + TILE * ROWLEN, 0, 1);    // global_prefetch_b8 (next tile)
  for (int idx = lane; idx < tot; idx += 32) ld[idx] = src[idx];
  __syncthreads();

  int n = n0 + lane;
  uint64_t key = ~0ull;                             // pad sorts last
  if (n < NANCH) {
    const float* r = ld + lane * ROWLEN;
    float obj = r[4];
    float best = -1e30f;
#pragma unroll 8
    for (int j = 0; j < NC; ++j) best = fmaxf(best, r[5 + j] * obj);
    bool valid = (obj > CONF_T) && (best > CONF_T);
    float score = valid ? best : -1.0f;
    uint32_t hi = ~map_score(score);                // asc key == desc score
    key = ((uint64_t)hi << 32) | (uint32_t)n;       // index asc tie-break
  }
  keys[(size_t)b * PADN + n0 + lane] = key;
}

// ---------------- Stage 2a: LDS-resident local sort (k = 2..2048) ----------
__global__ __launch_bounds__(1024)
void bitonic_local_sort(uint64_t* __restrict__ keys) {
  __shared__ uint64_t s[SORT_TILE];                 // 16 KB
  int b = blockIdx.x >> 4;                          // 16 tiles per batch
  int tile = blockIdx.x & 15;
  uint64_t* base = keys + (size_t)b * PADN + tile * SORT_TILE;
  int t = threadIdx.x;
  s[t] = base[t]; s[t + 1024] = base[t + 1024];
  int gbase = tile * SORT_TILE;                     // global index (direction bit)
  __syncthreads();
  for (int k = 2; k <= SORT_TILE; k <<= 1) {
    for (int j = k >> 1; j > 0; j >>= 1) {
      int l = ((t & ~(j - 1)) << 1) | (t & (j - 1));
      int p = l | j;
      bool up = (((gbase + l) & k) == 0);
      uint64_t a = s[l], c = s[p];
      if ((a > c) == up) { s[l] = c; s[p] = a; }
      __syncthreads();
    }
  }
  base[t] = s[t]; base[t + 1024] = s[t + 1024];
}

// ---------------- Stage 2b: global bitonic pass (stride >= 2048) -----------
__global__ void bitonic_global(uint64_t* __restrict__ keys, int j, int k) {
  int gid = blockIdx.x * blockDim.x + threadIdx.x;
  if (gid >= BATCH * PADN) return;
  int b = gid >> 15;
  int l = gid & (PADN - 1);
  int p = l ^ j;
  if (p <= l) return;
  uint64_t* base = keys + (size_t)b * PADN;
  uint64_t a = base[l], c = base[p];
  bool up = ((l & k) == 0);
  if ((a > c) == up) { base[l] = c; base[p] = a; }
}

// ---------------- Stage 2c: LDS-resident merge tail (stride 1024..1) -------
__global__ __launch_bounds__(1024)
void bitonic_local_merge(uint64_t* __restrict__ keys, int k) {
  __shared__ uint64_t s[SORT_TILE];
  int b = blockIdx.x >> 4;
  int tile = blockIdx.x & 15;
  uint64_t* base = keys + (size_t)b * PADN + tile * SORT_TILE;
  int t = threadIdx.x;
  s[t] = base[t]; s[t + 1024] = base[t + 1024];
  int gbase = tile * SORT_TILE;
  __syncthreads();
  for (int j = SORT_TILE >> 1; j > 0; j >>= 1) {
    int l = ((t & ~(j - 1)) << 1) | (t & (j - 1));
    int p = l | j;
    bool up = (((gbase + l) & k) == 0);
    uint64_t a = s[l], c = s[p];
    if ((a > c) == up) { s[l] = c; s[p] = a; }
    __syncthreads();
  }
  base[t] = s[t]; base[t + 1024] = s[t + 1024];
}

// ---------------- Stage 3: gather top-K, boxes, class, validity ------------
__global__ void gather_kernel(const float* __restrict__ pred,
                              const uint64_t* __restrict__ keys,
                              float* __restrict__ boxes, float* __restrict__ oboxes,
                              int* __restrict__ cls_t, int* __restrict__ top_i,
                              float* __restrict__ top_s, unsigned* __restrict__ valmask) {
  int gid = blockIdx.x * blockDim.x + threadIdx.x;   // BATCH*KTOP
  if (gid >= BATCH * KTOP) return;
  int b  = gid >> 10;
  uint64_t key = keys[(size_t)b * PADN + (gid & (KTOP - 1))];
  int idx = (int)(uint32_t)(key & 0xFFFFFFFFu);
  float score = unmap_score(~(uint32_t)(key >> 32));
  const float* row = pred + ((size_t)b * NANCH + idx) * ROWLEN;
  float cx = row[0], cy = row[1], w = row[2], h = row[3];
  float obj = row[4];
  float best = -1e30f; int cls = 0;
  for (int j = 0; j < NC; ++j) {           // first-max tie rule == jnp.argmax
    float v = row[5 + j] * obj;
    if (v > best) { best = v; cls = j; }
  }
  float x1 = cx - w * 0.5f, y1 = cy - h * 0.5f;
  float x2 = cx + w * 0.5f, y2 = cy + h * 0.5f;
  float off = (float)cls * MAX_WH;
  size_t o4 = (size_t)gid * 4;
  boxes[o4+0]=x1;      boxes[o4+1]=y1;      boxes[o4+2]=x2;      boxes[o4+3]=y2;
  oboxes[o4+0]=x1+off; oboxes[o4+1]=y1+off; oboxes[o4+2]=x2+off; oboxes[o4+3]=y2+off;
  cls_t[gid] = cls; top_i[gid] = idx; top_s[gid] = score;
  unsigned bal = (unsigned)__ballot(score > 0.0f);   // wave32 pack
  if ((threadIdx.x & 31) == 0) valmask[gid >> 5] = bal;
}

// ---------------- Stage 4: IoU suppression bitmask (wave32 ballot) ---------
__global__ void iou_mask_kernel(const float* __restrict__ oboxes,
                                unsigned* __restrict__ mask) {
  long long gid = (long long)blockIdx.x * blockDim.x + threadIdx.x; // B*K*K
  if (gid >= (long long)BATCH * KTOP * KTOP) return;
  int j = (int)(gid & (KTOP - 1));
  int i = (int)((gid >> 10) & (KTOP - 1));
  int b = (int)(gid >> 20);
  const float* A  = oboxes + (((size_t)b * KTOP + i) << 2);
  const float* Bb = oboxes + (((size_t)b * KTOP + j) << 2);
  float ax1=A[0],  ay1=A[1],  ax2=A[2],  ay2=A[3];
  float bx1=Bb[0], by1=Bb[1], bx2=Bb[2], by2=Bb[3];
  float areaA = (ax2-ax1)*(ay2-ay1);
  float areaB = (bx2-bx1)*(by2-by1);
  float iw = fmaxf(fminf(ax2,bx2) - fmaxf(ax1,bx1), 0.0f);
  float ih = fmaxf(fminf(ay2,by2) - fmaxf(ay1,by1), 0.0f);
  float inter = iw * ih;
  float iou = inter / (areaA + areaB - inter + 1e-9f);
  bool bit = (iou > IOU_T) && (j > i);
  unsigned bal = (unsigned)__ballot(bit);
  if ((threadIdx.x & 31) == 0) mask[(size_t)(gid >> 5)] = bal;
}

// ---------------- Stage 5: greedy NMS, one wave/batch ----------------------
__global__ void nms_kernel(const unsigned* __restrict__ mask,
                           const unsigned* __restrict__ valmask,
                           unsigned* __restrict__ keepmask) {
  int b = blockIdx.x;
  int lane = threadIdx.x;            // blockDim = 32
  __shared__ unsigned sup[32];
  __shared__ unsigned vs[32];
  sup[lane] = 0u;
  vs[lane]  = valmask[b * 32 + lane];
  unsigned keepW = 0u;
  __syncthreads();
  for (int i = 0; i < KTOP; ++i) {
    int w = i >> 5, bpos = i & 31;
    bool k = (((vs[w] >> bpos) & 1u) != 0u) && (((sup[w] >> bpos) & 1u) == 0u);
    __syncthreads();
    if (k) {
      sup[lane] |= mask[(((size_t)b * KTOP + i) << 5) + lane];
      if (w == lane) keepW |= (1u << bpos);
    }
    __syncthreads();
  }
  keepmask[b * 32 + lane] = keepW;
}

// ---------------- Stage 6: stable kept-first ordering via WMMA prefix scan -
// Inclusive prefix of 0/1 flags over 16-wide segments: D = A x U with
// V_WMMA_F32_16X16X32_F16 (exact: 0/1 f16 operands, f32 accumulate),
// then a 64-segment scalar carry pass.
__global__ void order_wmma_kernel(const unsigned* __restrict__ keepmask,
                                  int* __restrict__ order) {
  int b = blockIdx.x;
  int lane = threadIdx.x;            // blockDim = 32, full wave, EXEC all ones
  __shared__ unsigned km[32];
  __shared__ float sArr[KTOP];
  __shared__ float offs[64];
  __shared__ float totS;
  km[lane] = keepmask[b * 32 + lane];
  __syncthreads();

  // B matrix: U[k][n] = (k <= n), rows K=0..15 only (K=16..31 zero-padded).
  v16h bm;
#pragma unroll
  for (int hh = 0; hh < 16; ++hh)
    bm[hh] = (hh < 8 && (8 * (lane >> 4) + hh) <= (lane & 15)) ? (_Float16)1.0f : (_Float16)0.0f;

  int row   = lane & 15;             // A-matrix M
  int cbase = (lane < 16) ? 0 : 8;   // lanes 0-15: K=0..7 ; lanes 16-31: K=8..15
  int Ncol  = lane & 15;             // C/D column
  int Mbase = (lane < 16) ? 0 : 8;   // C/D row base

#pragma unroll
  for (int t = 0; t < 4; ++t) {
    v16h a;
    int base = t * 256 + row * 16 + cbase;
#pragma unroll
    for (int hh = 0; hh < 8; ++hh) {
      int i = base + hh;
      a[hh] = (_Float16)(float)((km[i >> 5] >> (i & 31)) & 1u);
    }
#pragma unroll
    for (int hh = 8; hh < 16; ++hh) a[hh] = (_Float16)0.0f;  // K=16..31 pad
    v8f c = {};
    c = __builtin_amdgcn_wmma_f32_16x16x32_f16(
        /*neg_a=*/false, a, /*neg_b=*/false, bm,
        /*c_mod=*/(short)0, c, /*reuse_a=*/false, /*reuse_b=*/false);
#pragma unroll
    for (int r = 0; r < 8; ++r)
      sArr[t * 256 + (Mbase + r) * 16 + Ncol] = c[r];   // segment-local prefix
  }
  __syncthreads();

  if (lane == 0) {                    // 64-segment exclusive carry
    float run = 0.0f;
    for (int seg = 0; seg < 64; ++seg) { offs[seg] = run; run += sArr[seg * 16 + 15]; }
    totS = run;
  }
  __syncthreads();

  int tot = (int)(totS + 0.5f);
  for (int i = lane; i < KTOP; i += 32) {
    float s = sArr[i] + offs[i >> 4];          // inclusive #kept up to i
    int si = (int)(s + 0.5f);
    bool kf = ((km[i >> 5] >> (i & 31)) & 1u) != 0u;
    int pos = kf ? (si - 1) : (tot + i - si);  // stable partition: kept first
    order[b * KTOP + pos] = i;
  }
}

// ---------------- Stage 7: finalize dets + logits gather -------------------
__global__ void finalize_dets_kernel(const float* __restrict__ boxes,
                                     const float* __restrict__ top_s,
                                     const int* __restrict__ cls_t,
                                     const int* __restrict__ order,
                                     const unsigned* __restrict__ keepmask,
                                     float* __restrict__ out) {
  int gid = blockIdx.x * blockDim.x + threadIdx.x;   // BATCH*MAXDET
  if (gid >= BATCH * MAXDET) return;
  int b = gid / MAXDET, d = gid - b * MAXDET;
  int i = order[b * KTOP + d];
  float kf = ((keepmask[b * 32 + (i >> 5)] >> (i & 31)) & 1u) ? 1.0f : 0.0f;
  size_t src = ((size_t)b * KTOP + i) << 2;
  float x1 = fmaxf(boxes[src + 0], 0.0f);
  float y1 = fmaxf(boxes[src + 1], 0.0f);
  float x2 = fminf(boxes[src + 2], IMGW);
  float y2 = fminf(boxes[src + 3], IMGH);
  float cf = top_s[b * KTOP + i];
  float cl = (float)cls_t[b * KTOP + i];
  float* o = out + (size_t)gid * 6;
  o[0] = x1 * kf; o[1] = y1 * kf; o[2] = x2 * kf;
  o[3] = y2 * kf; o[4] = cf * kf; o[5] = cl * kf;
}

__global__ void finalize_logits_kernel(const float* __restrict__ logits,
                                       const int* __restrict__ order,
                                       const int* __restrict__ top_i,
                                       const unsigned* __restrict__ keepmask,
                                       float* __restrict__ out) {
  int gid = blockIdx.x * blockDim.x + threadIdx.x;   // BATCH*MAXDET*NC
  if (gid >= BATCH * MAXDET * NC) return;
  int c = gid % NC;
  int d = (gid / NC) % MAXDET;
  int b = gid / (NC * MAXDET);
  int i = order[b * KTOP + d];
  float kf = ((keepmask[b * 32 + (i >> 5)] >> (i & 31)) & 1u) ? 1.0f : 0.0f;
  int idx = top_i[b * KTOP + i];
  out[gid] = logits[((size_t)b * NANCH + idx) * NC + c] * kf;
}

// ---------------------------------------------------------------------------
extern "C" void kernel_launch(void* const* d_in, const int* in_sizes, int n_in,
                              void* d_out, int out_size, void* d_ws, size_t ws_size,
                              hipStream_t stream) {
  const float* pred   = (const float*)d_in[0];
  const float* logits = (const float*)d_in[1];
  float* out = (float*)d_out;

  // workspace carve-up (~6.8 MB, L2-resident)
  char* ws = (char*)d_ws;
  size_t off = 0;
  uint64_t* keys  = (uint64_t*)(ws + off); off += (size_t)BATCH * PADN * sizeof(uint64_t);
  float* boxes    = (float*)(ws + off);    off += (size_t)BATCH * KTOP * 4 * sizeof(float);
  float* oboxes   = (float*)(ws + off);    off += (size_t)BATCH * KTOP * 4 * sizeof(float);
  int* cls_t      = (int*)(ws + off);      off += (size_t)BATCH * KTOP * sizeof(int);
  int* top_i      = (int*)(ws + off);      off += (size_t)BATCH * KTOP * sizeof(int);
  float* top_s    = (float*)(ws + off);    off += (size_t)BATCH * KTOP * sizeof(float);
  int* order      = (int*)(ws + off);      off += (size_t)BATCH * KTOP * sizeof(int);
  unsigned* valm  = (unsigned*)(ws + off); off += (size_t)BATCH * 32 * sizeof(unsigned);
  unsigned* keepm = (unsigned*)(ws + off); off += (size_t)BATCH * 32 * sizeof(unsigned);
  unsigned* mask  = (unsigned*)(ws + off); off += (size_t)BATCH * KTOP * 32 * sizeof(unsigned);

  // Stage 1: 16384 waves (1024 per batch), 4 waves/block
  score_key_kernel<<<BATCH * 1024 / 4, 128, 0, stream>>>(pred, keys);

  // Stage 2: hybrid bitonic sort, 15 launches instead of 120
  const int total = BATCH * PADN;
  bitonic_local_sort<<<BATCH * (PADN / SORT_TILE), 1024, 0, stream>>>(keys);
  for (int k = SORT_TILE * 2; k <= PADN; k <<= 1) {
    for (int j = k >> 1; j >= SORT_TILE; j >>= 1)
      bitonic_global<<<(total + 255) / 256, 256, 0, stream>>>(keys, j, k);
    bitonic_local_merge<<<BATCH * (PADN / SORT_TILE), 1024, 0, stream>>>(keys, k);
  }

  gather_kernel<<<(BATCH * KTOP + 255) / 256, 256, 0, stream>>>(
      pred, keys, boxes, oboxes, cls_t, top_i, top_s, valm);

  long long iou_total = (long long)BATCH * KTOP * KTOP;
  iou_mask_kernel<<<(unsigned)((iou_total + 255) / 256), 256, 0, stream>>>(oboxes, mask);

  nms_kernel<<<BATCH, 32, 0, stream>>>(mask, valm, keepm);
  order_wmma_kernel<<<BATCH, 32, 0, stream>>>(keepm, order);

  finalize_dets_kernel<<<(BATCH * MAXDET + 255) / 256, 256, 0, stream>>>(
      boxes, top_s, cls_t, order, keepm, out);
  finalize_logits_kernel<<<(BATCH * MAXDET * NC + 255) / 256, 256, 0, stream>>>(
      logits, order, top_i, keepm, out + (size_t)BATCH * MAXDET * 6);
}
